// PointNet2Backbone_56916906606964
// MI455X (gfx1250) — compile-verified, matrix operand order checked by reading, not run on hardware
//
#include <hip/hip_runtime.h>
#include <hip/hip_bf16.h>

// ---------------------------------------------------------------------------
// PointNet++ backbone for MI455X (gfx1250, wave32, WMMA).
// MLP matmuls: v_wmma_f32_16x16x32_f16, weights staged in LDS via
// double-buffered async global->LDS loads (ASYNCcnt), fused BN+ReLU epilogue.
// ---------------------------------------------------------------------------

typedef _Float16 f16;
typedef _Float16 h2   __attribute__((ext_vector_type(2)));
typedef _Float16 v8h  __attribute__((ext_vector_type(8)));
typedef __attribute__((ext_vector_type(16))) _Float16 v16h;
typedef __attribute__((ext_vector_type(8)))  float    v8f;

// ---------------------------- small utility kernels ------------------------

__global__ void f32_to_f16_kernel(const float* __restrict__ src, f16* __restrict__ dst, int n) {
    int i = blockIdx.x * blockDim.x + threadIdx.x;
    if (i < n) dst[i] = (f16)src[i];
}

// Convert layer weight W[K,N] (row-major, f32) into transposed padded
// Wt[N,Kpad] (f16) so B-tiles stream as contiguous K runs.
__global__ void wcvt_kernel(const float* __restrict__ W, f16* __restrict__ Wt,
                            int K, int N, int Kpad) {
    int i = blockIdx.x * blockDim.x + threadIdx.x;
    if (i >= N * Kpad) return;
    int n = i / Kpad, kk = i % Kpad;
    Wt[i] = (kk < K) ? (f16)W[(size_t)kk * N + n] : (f16)0;
}

// ------------------------------- FPS ---------------------------------------
__global__ __launch_bounds__(1024)
void fps_kernel(const float* __restrict__ pos, int N, int m, int* __restrict__ idx_out) {
    __shared__ float mind[2048];
    __shared__ float rv[1024];
    __shared__ int   ri[1024];
    __shared__ int   cur;
    int tid = threadIdx.x;
    float p0x = pos[0], p0y = pos[1], p0z = pos[2];
    for (int n = tid; n < N; n += blockDim.x) {
        float dx = pos[n*3]-p0x, dy = pos[n*3+1]-p0y, dz = pos[n*3+2]-p0z;
        mind[n] = dx*dx + dy*dy + dz*dz;
    }
    if (tid == 0) idx_out[0] = 0;
    __syncthreads();
    for (int i = 1; i < m; ++i) {
        float bv = -1.0f; int bi = 0x7fffffff;
        for (int n = tid; n < N; n += blockDim.x)
            if (mind[n] > bv) { bv = mind[n]; bi = n; }
        rv[tid] = bv; ri[tid] = bi;
        __syncthreads();
        for (int s = blockDim.x >> 1; s > 0; s >>= 1) {
            if (tid < s) {
                float ov = rv[tid+s]; int oi = ri[tid+s];
                if (ov > rv[tid] || (ov == rv[tid] && oi < ri[tid])) { rv[tid] = ov; ri[tid] = oi; }
            }
            __syncthreads();
        }
        if (tid == 0) { cur = ri[0]; idx_out[i] = cur; }
        __syncthreads();
        int c = cur;
        float cx = pos[c*3], cy = pos[c*3+1], cz = pos[c*3+2];
        for (int n = tid; n < N; n += blockDim.x) {
            float dx = pos[n*3]-cx, dy = pos[n*3+1]-cy, dz = pos[n*3+2]-cz;
            mind[n] = fminf(mind[n], dx*dx + dy*dy + dz*dz);
        }
        __syncthreads();
    }
}

__global__ void gather_pos_kernel(const float* __restrict__ pos, const int* __restrict__ idx,
                                  float* __restrict__ posd, int m) {
    int i = blockIdx.x * blockDim.x + threadIdx.x;
    if (i >= m) return;
    int j = idx[i];
    posd[i*3]   = pos[j*3];
    posd[i*3+1] = pos[j*3+1];
    posd[i*3+2] = pos[j*3+2];
}

// First-K-by-index radius neighbors (PyG-style cap).
__global__ void radius_kernel(const float* __restrict__ src, int Ns,
                              const float* __restrict__ dst, int m, float r2,
                              int* __restrict__ nbr, int* __restrict__ cnt, int K) {
    int i = blockIdx.x * blockDim.x + threadIdx.x;
    if (i >= m) return;
    float dx = dst[i*3], dy = dst[i*3+1], dz = dst[i*3+2];
    int c = 0;
    for (int j = 0; j < Ns && c < K; ++j) {
        float ax = src[j*3]-dx, ay = src[j*3+1]-dy, az = src[j*3+2]-dz;
        float d2 = ax*ax + ay*ay + az*az;
        if (d2 <= r2) { nbr[i*K + c] = j; ++c; }
    }
    cnt[i] = c;
    for (int t = c; t < K; ++t) nbr[i*K + t] = 0;
}

__global__ void gather_feat_kernel(const f16* __restrict__ xsrc, int C,
                                   const float* __restrict__ pos, const float* __restrict__ posd,
                                   const int* __restrict__ nbr, const int* __restrict__ cnt,
                                   int m, int Kpad, f16* __restrict__ X) {
    const int K = 64;
    int r = blockIdx.x * blockDim.x + threadIdx.x;
    if (r >= m * K) return;
    int i = r / K, kk = r % K;
    f16* xr = X + (size_t)r * Kpad;
    if (kk >= cnt[i]) { for (int c = 0; c < Kpad; ++c) xr[c] = (f16)0; return; }
    int j = nbr[r];
    const f16* fr = xsrc + (size_t)j * C;
    for (int c = 0; c < C; ++c) xr[c] = fr[c];
    for (int d = 0; d < 3; ++d) xr[C+d] = (f16)(pos[j*3+d] - posd[i*3+d]);
    for (int c = C + 3; c < Kpad; ++c) xr[c] = (f16)0;
}

__global__ void aggregate_max_kernel(const f16* __restrict__ msg, const int* __restrict__ cnt,
                                     int m, int K, int C, f16* __restrict__ xo) {
    int t = blockIdx.x * blockDim.x + threadIdx.x;
    if (t >= m * C) return;
    int i = t / C, c = t % C;
    int n = cnt[i];
    float mx = -3.0e38f;
    for (int kk = 0; kk < n; ++kk)
        mx = fmaxf(mx, (float)msg[((size_t)(i*K + kk)) * C + c]);
    xo[(size_t)i * C + c] = (f16)mx;
}

__global__ void concat2_kernel(const f16* __restrict__ A, int Ca, int strideA,
                               const f16* __restrict__ Bm, int Cb,
                               int M, int Kpad, f16* __restrict__ X) {
    int i = blockIdx.x * blockDim.x + threadIdx.x;
    if (i >= M) return;
    f16* xr = X + (size_t)i * Kpad;
    const f16* ar = A + (size_t)i * strideA;
    const f16* br = Bm + (size_t)i * Cb;
    for (int c = 0; c < Kpad; ++c) {
        f16 v = (f16)0;
        if (c < Ca) v = ar[c];
        else if (c < Ca + Cb) v = br[c - Ca];
        xr[c] = v;
    }
}

__global__ void global_max_kernel(const f16* __restrict__ Y, int M, int C, f16* __restrict__ g) {
    int c = blockIdx.x * blockDim.x + threadIdx.x;
    if (c >= C) return;
    float mx = -3.0e38f;
    for (int i = 0; i < M; ++i) mx = fmaxf(mx, (float)Y[(size_t)i * C + c]);
    g[c] = (f16)mx;
}

__global__ void knn3_kernel(const f16* __restrict__ xc, const float* __restrict__ pc,
                            const float* __restrict__ pf, int Nc, int Nf, int C,
                            f16* __restrict__ out) {
    int f = blockIdx.x * blockDim.x + threadIdx.x;
    if (f >= Nf) return;
    float px = pf[f*3], py = pf[f*3+1], pz = pf[f*3+2];
    int bi[3] = {-1, -1, -1};
    float bd[3];
    for (int t = 0; t < 3; ++t) {
        float best = 3.0e38f; int besti = 0;
        for (int j = 0; j < Nc; ++j) {
            if (j == bi[0] || j == bi[1]) continue;
            float dx = pc[j*3]-px, dy = pc[j*3+1]-py, dz = pc[j*3+2]-pz;
            float d2 = dx*dx + dy*dy + dz*dz;
            if (d2 < best) { best = d2; besti = j; }
        }
        bd[t] = best; bi[t] = besti;
    }
    float w0 = 1.0f / fmaxf(bd[0], 1e-16f);
    float w1 = 1.0f / fmaxf(bd[1], 1e-16f);
    float w2 = 1.0f / fmaxf(bd[2], 1e-16f);
    float ws = w0 + w1 + w2;
    const f16* r0 = xc + (size_t)bi[0] * C;
    const f16* r1 = xc + (size_t)bi[1] * C;
    const f16* r2 = xc + (size_t)bi[2] * C;
    f16* o = out + (size_t)f * C;
    for (int c = 0; c < C; ++c)
        o[c] = (f16)((w0*(float)r0[c] + w1*(float)r1[c] + w2*(float)r2[c]) / ws);
}

// ---------------------- WMMA GEMM + BN + ReLU ------------------------------
// Y = relu(gamma*(X@W + bias) + beta).  X:[M,Kpad] f16, Wt:[N,Kpad] f16 (W^T).
// Block = 128 threads (4 waves). Block tile = 64(M) x 64(N); wave w owns rows
// [w*16, w*16+16) across all 64 cols -> 4 accumulators per wave, A-fragment
// reused 4x. The shared 64x32 B-tile (4 KB) is staged in LDS with
// double-buffered global_load_async_to_lds_b128 (ASYNCcnt pipeline), then fed
// to v_wmma via ds_load_b128. M, N multiples of 64; Kpad multiple of 32.
__global__ __launch_bounds__(128)
void gemm_bn_relu_kernel(const f16* __restrict__ X, const f16* __restrict__ Wt,
                         const float* __restrict__ bias, const float* __restrict__ gamma,
                         const float* __restrict__ beta,
                         f16* __restrict__ Yh, float* __restrict__ Yf,
                         int N, int Kpad) {
    __shared__ f16 Bt[2][64 * 32];   // [buf][col*32 + k], 64B per column

    const int tid   = threadIdx.x;
    const int lane  = tid & 31;
    const int wave  = tid >> 5;
    const int g     = lane >> 4;       // lane group 0/1
    const int mr    = lane & 15;
    const int tileM = blockIdx.y * 64 + wave * 16;
    const int colBlk= blockIdx.x * 64;

    const f16* Arow = X + (size_t)(tileM + mr) * Kpad;

    // Cooperative async staging of one 64x32 B-tile: 256 x b128, 2 per thread.
    // seg -> col = seg>>2, part = seg&3 (part = 8-half chunk within the k-run).
    const int seg0 = tid * 2;
    const int c0 = seg0 >> 2,       p0 = seg0 & 3;
    const int c1 = (seg0 + 1) >> 2, p1 = (seg0 + 1) & 3;
    const f16* gB0 = Wt + (size_t)(colBlk + c0) * Kpad + p0 * 8;
    const f16* gB1 = Wt + (size_t)(colBlk + c1) * Kpad + p1 * 8;
    const unsigned ldsOff0a = (unsigned)(unsigned long long)&Bt[0][c0 * 32 + p0 * 8];
    const unsigned ldsOff1a = (unsigned)(unsigned long long)&Bt[0][c1 * 32 + p1 * 8];
    const unsigned bufStride = (unsigned)(64 * 32 * sizeof(f16));

    auto issueTile = [&](int k0, int buf) {
        unsigned long long ga0 = (unsigned long long)(gB0 + k0);
        unsigned long long ga1 = (unsigned long long)(gB1 + k0);
        unsigned l0 = ldsOff0a + (unsigned)buf * bufStride;
        unsigned l1 = ldsOff1a + (unsigned)buf * bufStride;
        asm volatile("global_load_async_to_lds_b128 %0, %1, off"
                     :: "v"(l0), "v"(ga0) : "memory");
        asm volatile("global_load_async_to_lds_b128 %0, %1, off"
                     :: "v"(l1), "v"(ga1) : "memory");
    };

    v8f acc[4] = {};
    const int nk = Kpad >> 5;

    issueTile(0, 0);
    for (int kt = 0; kt < nk; ++kt) {
        const int p = kt & 1;
        if (kt + 1 < nk) {
            issueTile((kt + 1) << 5, p ^ 1);
            // tile kt's 2 async ops are older than the 2 just issued; async
            // loads complete in order, so draining to <=2 lands tile kt.
            asm volatile("s_wait_asynccnt 2" ::: "memory");
        } else {
            asm volatile("s_wait_asynccnt 0" ::: "memory");
        }
        __syncthreads();   // all waves' portions of tile kt visible in LDS

        // A fragment (global): halves [8g..8g+7] -> a[0..7], [16+8g..] -> a[8..15]
        const f16* ap = Arow + (kt << 5);
        if (kt + 1 < nk) __builtin_prefetch((const void*)(ap + 32), 0, 1);
        v8h alo = *reinterpret_cast<const v8h*>(ap + g * 8);
        v8h ahi = *reinterpret_cast<const v8h*>(ap + 16 + g * 8);
        v16h a;
#pragma unroll
        for (int i = 0; i < 8; ++i) { a[i] = alo[i]; a[i + 8] = ahi[i]; }

        const f16* Bb = &Bt[p][0];
#pragma unroll
        for (int nt = 0; nt < 4; ++nt) {
            const int col = nt * 16 + mr;
            v8h blo = *reinterpret_cast<const v8h*>(Bb + col * 32 + g * 8);
            v8h bhi = *reinterpret_cast<const v8h*>(Bb + col * 32 + 16 + g * 8);
            v16h b;
#pragma unroll
            for (int i = 0; i < 8; ++i) { b[i] = blo[i]; b[i + 8] = bhi[i]; }
            acc[nt] = __builtin_amdgcn_wmma_f32_16x16x32_f16(
                false, a, false, b, (short)0, acc[nt], false, false);
        }
        __syncthreads();   // tile kt fully consumed before its buffer is reused
    }

#pragma unroll
    for (int nt = 0; nt < 4; ++nt) {
        const int col = colBlk + nt * 16 + mr;
        const float gm = gamma[col], bs = bias[col], bt = beta[col];
#pragma unroll
        for (int i = 0; i < 8; ++i) {
            int row = tileM + i + g * 8;
            float z = gm * (acc[nt][i] + bs) + bt;
            z = fmaxf(z, 0.0f);
            size_t o = (size_t)row * N + col;
            Yh[o] = (f16)z;
            if (Yf) Yf[o] = z;
        }
    }
}

// ------------------------------- host side ---------------------------------

static inline void gemm(hipStream_t s, const f16* X, const f16* Wt,
                        const float* b, const float* g, const float* bt,
                        f16* Yh, float* Yf, int M, int N, int Kpad) {
    dim3 grid(N / 64, M / 64);
    gemm_bn_relu_kernel<<<grid, 128, 0, s>>>(X, Wt, b, g, bt, Yh, Yf, N, Kpad);
}

extern "C" void kernel_launch(void* const* d_in, const int* in_sizes, int n_in,
                              void* d_out, int out_size, void* d_ws, size_t ws_size,
                              hipStream_t stream) {
    (void)in_sizes; (void)n_in; (void)out_size; (void)ws_size;

    const float* x   = (const float*)d_in[0];   // [16384,64]
    const float* pos = (const float*)d_in[1];   // [16384,3]

    struct LayerDef { int K, N, Kpad, widx; };
    static const LayerDef Ld[15] = {
        { 67,   64,   96,  3}, { 64,  64,  64,  7}, { 64,  128,  64, 11},   // sa1
        {131,   64,  160, 15}, { 64,  64,  64, 19}, { 64,  256,  64, 23},   // sa2
        {259,  256,  288, 27}, {256, 512, 256, 31}, {512, 1024, 512, 35},   // sa3
        {1280, 256, 1280, 39}, {256, 256, 256, 43},                          // fp3
        {384,  128,  384, 47}, {128, 128, 128, 51},                          // fp2
        {192,  128,  192, 55}, {128, 128, 128, 59},                          // fp1
    };
    const float* LB[15]; const float* LG[15]; const float* LBT[15];
    for (int l = 0; l < 15; ++l) {
        LB[l]  = (const float*)d_in[Ld[l].widx + 1];
        LG[l]  = (const float*)d_in[Ld[l].widx + 2];
        LBT[l] = (const float*)d_in[Ld[l].widx + 3];
    }

    // ---- carve workspace (deterministic layout each call) ----
    char* wp = (char*)d_ws;
    auto take = [&wp](size_t bytes) -> char* {
        char* p = wp; wp += (bytes + 255) & ~(size_t)255; return p;
    };
    f16* Wt[15];
    for (int l = 0; l < 15; ++l)
        Wt[l] = (f16*)take((size_t)Ld[l].N * Ld[l].Kpad * sizeof(f16));
    f16*   xh   = (f16*)take((size_t)16384 * 64 * sizeof(f16));
    f16*   bufX = (f16*)take((size_t)65536 * 96  * sizeof(f16));
    f16*   bufA = (f16*)take((size_t)65536 * 128 * sizeof(f16));
    f16*   bufB = (f16*)take((size_t)65536 * 128 * sizeof(f16));
    f16*   x1   = (f16*)take((size_t)1024 * 128 * sizeof(f16));
    f16*   x2   = (f16*)take((size_t)256 * 256 * sizeof(f16));
    f16*   x2f  = (f16*)take((size_t)256 * 256 * sizeof(f16));
    f16*   x1f  = (f16*)take((size_t)1024 * 128 * sizeof(f16));
    f16*   gvec = (f16*)take((size_t)1024 * sizeof(f16));
    f16*   hbuf = (f16*)take((size_t)262144 * sizeof(f16));
    f16*   p2h  = (f16*)take((size_t)768 * sizeof(f16));
    float* pos1 = (float*)take((size_t)1024 * 3 * sizeof(float));
    float* pos2 = (float*)take((size_t)256 * 3 * sizeof(float));
    int*   idx1 = (int*)take((size_t)1024 * sizeof(int));
    int*   idx2 = (int*)take((size_t)256 * sizeof(int));
    int*   nbr1 = (int*)take((size_t)1024 * 64 * sizeof(int));
    int*   nbr2 = (int*)take((size_t)256 * 64 * sizeof(int));
    int*   cnt1 = (int*)take((size_t)1024 * sizeof(int));
    int*   cnt2 = (int*)take((size_t)256 * sizeof(int));

    // ---- weights: f32 -> f16 transposed+padded ----
    for (int l = 0; l < 15; ++l) {
        int n = Ld[l].N * Ld[l].Kpad;
        wcvt_kernel<<<(n + 255) / 256, 256, 0, stream>>>(
            (const float*)d_in[Ld[l].widx], Wt[l], Ld[l].K, Ld[l].N, Ld[l].Kpad);
    }
    f32_to_f16_kernel<<<(16384 * 64 + 255) / 256, 256, 0, stream>>>(x, xh, 16384 * 64);

    // ---- per-cloud pipeline ----
    for (int c = 0; c < 8; ++c) {
        const float* posc = pos + (size_t)c * 2048 * 3;
        const f16*   xhc  = xh  + (size_t)c * 2048 * 64;
        float*       outc = (float*)d_out + (size_t)c * 2048 * 128;

        // SA1: fps(2048->1024), r=0.2, MLP 67->64->64->128, max over K
        fps_kernel<<<1, 1024, 0, stream>>>(posc, 2048, 1024, idx1);
        gather_pos_kernel<<<4, 256, 0, stream>>>(posc, idx1, pos1, 1024);
        radius_kernel<<<4, 256, 0, stream>>>(posc, 2048, pos1, 1024, 0.04f, nbr1, cnt1, 64);
        gather_feat_kernel<<<256, 256, 0, stream>>>(xhc, 64, posc, pos1, nbr1, cnt1, 1024, 96, bufX);
        gemm(stream, bufX, Wt[0], LB[0], LG[0], LBT[0], bufA, nullptr, 65536, 64, 96);
        gemm(stream, bufA, Wt[1], LB[1], LG[1], LBT[1], bufB, nullptr, 65536, 64, 64);
        gemm(stream, bufB, Wt[2], LB[2], LG[2], LBT[2], bufA, nullptr, 65536, 128, 64);
        aggregate_max_kernel<<<(1024 * 128 + 255) / 256, 256, 0, stream>>>(bufA, cnt1, 1024, 64, 128, x1);

        // SA2: fps(1024->256), r=0.4, MLP 131->64->64->256
        fps_kernel<<<1, 1024, 0, stream>>>(pos1, 1024, 256, idx2);
        gather_pos_kernel<<<1, 256, 0, stream>>>(pos1, idx2, pos2, 256);
        radius_kernel<<<1, 256, 0, stream>>>(pos1, 1024, pos2, 256, 0.16f, nbr2, cnt2, 64);
        gather_feat_kernel<<<64, 256, 0, stream>>>(x1, 128, pos1, pos2, nbr2, cnt2, 256, 160, bufX);
        gemm(stream, bufX, Wt[3], LB[3], LG[3], LBT[3], bufA, nullptr, 16384, 64, 160);
        gemm(stream, bufA, Wt[4], LB[4], LG[4], LBT[4], bufB, nullptr, 16384, 64, 64);
        gemm(stream, bufB, Wt[5], LB[5], LG[5], LBT[5], bufA, nullptr, 16384, 256, 64);
        aggregate_max_kernel<<<(256 * 256 + 255) / 256, 256, 0, stream>>>(bufA, cnt2, 256, 64, 256, x2);

        // SA3 global: MLP([x2|pos2]) 259->256->512->1024, global max -> g
        f32_to_f16_kernel<<<3, 256, 0, stream>>>(pos2, p2h, 768);
        concat2_kernel<<<1, 256, 0, stream>>>(x2, 256, 256, p2h, 3, 256, 288, bufX);
        gemm(stream, bufX, Wt[6], LB[6], LG[6], LBT[6], bufA, nullptr, 256, 256, 288);
        gemm(stream, bufA, Wt[7], LB[7], LG[7], LBT[7], bufB, nullptr, 256, 512, 256);
        gemm(stream, bufB, Wt[8], LB[8], LG[8], LBT[8], bufA, nullptr, 256, 1024, 512);
        global_max_kernel<<<4, 256, 0, stream>>>(bufA, 256, 1024, gvec);

        // FP3: [broadcast g | x2] -> MLP 1280->256->256
        concat2_kernel<<<1, 256, 0, stream>>>(gvec, 1024, 0, x2, 256, 256, 1280, bufX);
        gemm(stream, bufX, Wt[9],  LB[9],  LG[9],  LBT[9],  bufA, nullptr, 256, 256, 1280);
        gemm(stream, bufA, Wt[10], LB[10], LG[10], LBT[10], x2f,  nullptr, 256, 256, 256);

        // FP2: knn3(x2f: pos2->pos1), [h | x1] -> MLP 384->128->128
        knn3_kernel<<<4, 256, 0, stream>>>(x2f, pos2, pos1, 256, 1024, 256, hbuf);
        concat2_kernel<<<4, 256, 0, stream>>>(hbuf, 256, 256, x1, 128, 1024, 384, bufX);
        gemm(stream, bufX, Wt[11], LB[11], LG[11], LBT[11], bufA, nullptr, 1024, 128, 384);
        gemm(stream, bufA, Wt[12], LB[12], LG[12], LBT[12], x1f,  nullptr, 1024, 128, 128);

        // FP1: knn3(x1f: pos1->pos), [h | x] -> MLP 192->128->128 -> f32 out
        knn3_kernel<<<8, 256, 0, stream>>>(x1f, pos1, posc, 1024, 2048, 128, hbuf);
        concat2_kernel<<<8, 256, 0, stream>>>(hbuf, 128, 128, xhc, 64, 2048, 192, bufX);
        gemm(stream, bufX, Wt[13], LB[13], LG[13], LBT[13], bufA, nullptr, 2048, 128, 192);
        gemm(stream, bufA, Wt[14], LB[14], LG[14], LBT[14], bufB, outc,    2048, 128, 128);
    }
}
